// attn_block_separate_cat_66357244723398
// MI455X (gfx1250) — matrix operand order repaired and checked by reading, not compiled, not run on hardware
//
#include <hip/hip_runtime.h>
#include <hip/hip_bf16.h>

typedef __bf16 bf16;
typedef __attribute__((ext_vector_type(16))) __bf16 v16bf;
typedef __attribute__((ext_vector_type(8)))  __bf16 v8bf;
typedef __attribute__((ext_vector_type(8)))  float  v8f;

#define Bn   8
#define Tn   2048
#define Dn   256
#define NROW (Bn * Tn)   // 16384

// Wave-internal LDS fence: DS ops from one wave are in-order at the LDS, so
// a dscnt wait + scheduling barrier makes cross-lane LDS communication safe
// without a workgroup barrier (we run 1 wave per workgroup).
__device__ __forceinline__ void wave_lds_fence() {
  __builtin_amdgcn_sched_barrier(0);
  asm volatile("s_wait_dscnt 0" ::: "memory");
  __builtin_amdgcn_sched_barrier(0);
}

// ---------------------------------------------------------------------------
// Kernel A1: transpose + bf16-convert the 6 weight matrices.
// For Wq/Wk/Wv (2D x D) only the top D rows are needed (aux half is folded
// into a per-batch bias by prep_auxbias).
// WT[mat][n*Dn + k] = bf16(W[k*Dn + n])
// ---------------------------------------------------------------------------
__global__ void prep_weights(const float* __restrict__ Wql, const float* __restrict__ Wq,
                             const float* __restrict__ Wk,  const float* __restrict__ Wv,
                             const float* __restrict__ Ws,  const float* __restrict__ Wt,
                             bf16* __restrict__ WT) {
  const int mat = blockIdx.y;   // 0..5
  const int n   = blockIdx.x;   // 0..255
  const int k   = threadIdx.x;  // 0..255
  const float* W;
  switch (mat) {
    case 0:  W = Wql; break;
    case 1:  W = Wq;  break;
    case 2:  W = Wk;  break;
    case 3:  W = Wv;  break;
    case 4:  W = Ws;  break;
    default: W = Wt;  break;
  }
  WT[(size_t)mat * Dn * Dn + (size_t)n * Dn + k] = (bf16)W[(size_t)k * Dn + n];
}

// ---------------------------------------------------------------------------
// Kernel A2: per-batch bias from the aux half of the concat GEMMs:
// auxb[which][b][n] = sum_j aux[b][j] * W[(Dn+j)*Dn + n] + bias[n]
// ---------------------------------------------------------------------------
__global__ void prep_auxbias(const float* __restrict__ aux,
                             const float* __restrict__ Wq, const float* __restrict__ bq,
                             const float* __restrict__ Wk, const float* __restrict__ bk,
                             const float* __restrict__ Wv, const float* __restrict__ bv,
                             float* __restrict__ auxb) {
  const int which = blockIdx.y;  // 0=q 1=k 2=v
  const int b = blockIdx.x;
  const int n = threadIdx.x;
  const float* W  = (which == 0) ? Wq : (which == 1) ? Wk : Wv;
  const float* bb = (which == 0) ? bq : (which == 1) ? bk : bv;
  float s = bb[n];
  for (int j = 0; j < Dn; ++j)
    s += aux[b * Dn + j] * W[(size_t)(Dn + j) * Dn + n];
  auxb[(size_t)(which * Bn + b) * Dn + n] = s;
}

// ---------------------------------------------------------------------------
// 16-row-stripe GEMM: acc[16 ntiles][8] = X(16x256 bf16, LDS) @ WT^T
// A-fragment: lane m=lane&15 -> row M=m; elem i -> K = g*16 + i
// B-fragment: WT row-major [N][K] -> one contiguous 32B load per lane.
// ---------------------------------------------------------------------------
__device__ __forceinline__ void wgemm(const bf16 (*X)[Dn], const bf16* __restrict__ WT,
                                      int lane, v8f acc[16]) {
  const int g = lane >> 4, m = lane & 15;
#pragma unroll
  for (int nt = 0; nt < 16; ++nt)
#pragma unroll
    for (int r = 0; r < 8; ++r) acc[nt][r] = 0.f;
  for (int ks = 0; ks < Dn; ks += 32) {
    v16bf a;
    const bf16* rowp = &X[m][ks + g * 16];
#pragma unroll
    for (int i = 0; i < 8; ++i) { a[i] = rowp[i]; a[i + 8] = rowp[8 + i]; }
#pragma unroll
    for (int nt = 0; nt < 16; ++nt) {
      v16bf bw = *(const v16bf*)(WT + (size_t)(nt * 16 + m) * Dn + ks + g * 16);
      acc[nt] = __builtin_amdgcn_wmma_f32_16x16x32_bf16(false, a, false, bw,
                                                        (short)0, acc[nt], false, false);
    }
  }
}

__device__ __forceinline__ void store_yf(const v8f acc[16], const float* __restrict__ biasc,
                                         float (*Yf)[Dn], int lane) {
  const int g = lane >> 4, m = lane & 15;
#pragma unroll
  for (int nt = 0; nt < 16; ++nt)
#pragma unroll
    for (int r = 0; r < 8; ++r)
      Yf[r + 8 * g][nt * 16 + m] = acc[nt][r] + biasc[nt * 16 + m];
}

// LayerNorm over each 256-wide row of Yf; two lanes per row (xor-16 combine).
__device__ __forceinline__ void ln_store(const float (*Yf)[Dn],
                                         const float* __restrict__ gamma,
                                         const float* __restrict__ beta,
                                         bf16* dst, int lane) {
  const int row = lane & 15, h = lane >> 4;
  float s = 0.f, s2 = 0.f;
  for (int c = h * 128; c < h * 128 + 128; ++c) {
    float v = Yf[row][c];
    s += v; s2 += v * v;
  }
  s  += __shfl_xor(s, 16, 32);
  s2 += __shfl_xor(s2, 16, 32);
  const float mean = s * (1.f / Dn);
  const float rstd = rsqrtf(s2 * (1.f / Dn) - mean * mean + 1e-5f);
  for (int c = h * 128; c < h * 128 + 128; ++c) {
    float v = (Yf[row][c] - mean) * rstd * gamma[c] + beta[c];
    dst[(size_t)row * Dn + c] = (bf16)v;
  }
}

// ---------------------------------------------------------------------------
// Kernel B: fused projections for a 16-row stripe (one wave per block).
//   q = LN(PReLU(  (query@Wql+bql) @ Wq_top + auxb_q ))
//   k = LN(PReLU(  value @ Wk_top + auxb_k ))
//   v = sigmoid(vLN@Ws+bs) * tanh(vLN@Wt+bt),  vLN = LN(PReLU(value@Wv_top+auxb_v))
// ---------------------------------------------------------------------------
__global__ __launch_bounds__(32) void proj_kernel(
    const float* __restrict__ query, const float* __restrict__ value,
    const bf16* __restrict__ WT, const float* __restrict__ auxb,
    const float* __restrict__ bql,
    const float* __restrict__ aqp, const float* __restrict__ gq, const float* __restrict__ btq,
    const float* __restrict__ akp, const float* __restrict__ gk, const float* __restrict__ btk,
    const float* __restrict__ avp, const float* __restrict__ gv, const float* __restrict__ btv,
    const float* __restrict__ bs, const float* __restrict__ btw,
    bf16* __restrict__ qb, bf16* __restrict__ kb, bf16* __restrict__ vb) {
  __shared__ bf16  X[16][Dn];    // bf16 GEMM input stripe
  __shared__ float Yf[16][Dn];   // f32 GEMM output / LN input
  __shared__ bf16  Sg[16][Dn];   // sigmoid(s) for v-gating

  const int lane = threadIdx.x;
  const int g = lane >> 4, m = lane & 15;
  const int row0 = blockIdx.x * 16;
  const int b = row0 / Tn;
  const bf16* WqlT = WT;
  const bf16* WqT  = WT + 1 * 65536;
  const bf16* WkT  = WT + 2 * 65536;
  const bf16* WvT  = WT + 3 * 65536;
  const bf16* WsT  = WT + 4 * 65536;
  const bf16* WtT  = WT + 5 * 65536;

  v8f acc[16];

  // ------------------ q path ------------------
  for (int idx = lane; idx < 16 * Dn; idx += 32) {
    int rr = idx >> 8, cc = idx & 255;
    X[rr][cc] = (bf16)query[(size_t)(row0 + rr) * Dn + cc];
  }
  wave_lds_fence();
  wgemm(X, WqlT, lane, acc);
  store_yf(acc, bql, Yf, lane);
  wave_lds_fence();
  for (int idx = lane; idx < 16 * Dn; idx += 32) {
    int rr = idx >> 8, cc = idx & 255;
    X[rr][cc] = (bf16)Yf[rr][cc];
  }
  wave_lds_fence();
  wgemm(X, WqT, lane, acc);
  store_yf(acc, auxb + (size_t)b * Dn, Yf, lane);
  wave_lds_fence();
  {
    const float aq = aqp[0];
    for (int idx = lane; idx < 16 * Dn; idx += 32) {
      int rr = idx >> 8, cc = idx & 255;
      float v = Yf[rr][cc];
      Yf[rr][cc] = v >= 0.f ? v : aq * v;
    }
  }
  wave_lds_fence();
  ln_store(Yf, gq, btq, qb + (size_t)row0 * Dn, lane);

  // ------------------ k path ------------------
  wave_lds_fence();
  for (int idx = lane; idx < 16 * Dn; idx += 32) {
    int rr = idx >> 8, cc = idx & 255;
    X[rr][cc] = (bf16)value[(size_t)(row0 + rr) * Dn + cc];
  }
  wave_lds_fence();
  wgemm(X, WkT, lane, acc);
  store_yf(acc, auxb + (size_t)(Bn + b) * Dn, Yf, lane);
  wave_lds_fence();
  {
    const float ak = akp[0];
    for (int idx = lane; idx < 16 * Dn; idx += 32) {
      int rr = idx >> 8, cc = idx & 255;
      float v = Yf[rr][cc];
      Yf[rr][cc] = v >= 0.f ? v : ak * v;
    }
  }
  wave_lds_fence();
  ln_store(Yf, gk, btk, kb + (size_t)row0 * Dn, lane);

  // ------------------ v path ------------------  (X still holds bf16(value))
  wave_lds_fence();
  wgemm(X, WvT, lane, acc);
  store_yf(acc, auxb + (size_t)(2 * Bn + b) * Dn, Yf, lane);
  wave_lds_fence();
  {
    const float av = avp[0];
    for (int idx = lane; idx < 16 * Dn; idx += 32) {
      int rr = idx >> 8, cc = idx & 255;
      float v = Yf[rr][cc];
      Yf[rr][cc] = v >= 0.f ? v : av * v;
    }
  }
  wave_lds_fence();
  ln_store(Yf, gv, btv, &X[0][0], lane);   // vLN (bf16) back into LDS X
  wave_lds_fence();
  wgemm(X, WsT, lane, acc);
  store_yf(acc, bs, Yf, lane);
  wave_lds_fence();
  for (int idx = lane; idx < 16 * Dn; idx += 32) {
    int rr = idx >> 8, cc = idx & 255;
    Sg[rr][cc] = (bf16)(1.f / (1.f + __expf(-Yf[rr][cc])));
  }
  wave_lds_fence();
  wgemm(X, WtT, lane, acc);
#pragma unroll
  for (int nt = 0; nt < 16; ++nt)
#pragma unroll
    for (int r = 0; r < 8; ++r) {
      float tv = tanhf(acc[nt][r] + btw[nt * 16 + m]);
      vb[(size_t)(row0 + r + 8 * g) * Dn + nt * 16 + m] =
          (bf16)((float)Sg[r + 8 * g][nt * 16 + m] * tv);
    }
}

// ---------------------------------------------------------------------------
// Kernel C: flash attention over one 16-row q tile (one wave per block).
// Processes TWO 16-col s-tiles per iteration so the P@V WMMA uses a full
// K=32 A-fragment (no zero padding, no idle lanes on V loads).
// Causal + exact-zero quirk + col<len mask; online softmax in f32.
// ---------------------------------------------------------------------------
__global__ __launch_bounds__(32) void attn_kernel(
    const bf16* __restrict__ qb, const bf16* __restrict__ kb, const bf16* __restrict__ vb,
    const int* __restrict__ input_len, float* __restrict__ out) {
  __shared__ bf16 sP[16][32];   // P pair-tile staging (C-layout -> A-layout)

  const int lane = threadIdx.x;
  const int g = lane >> 4, m = lane & 15;
  const int row0 = blockIdx.x * 16;
  const int b  = row0 / Tn;
  const int t0 = row0 % Tn;
  const int jt = t0 >> 4;       // last causal 16-wide s-tile
  const int len = input_len[b];

  // Preload q A-fragments for all 8 K-steps (K=256).
  v16bf qa[8];
#pragma unroll
  for (int ks = 0; ks < 8; ++ks) {
    const bf16* p = qb + (size_t)(row0 + m) * Dn + ks * 32 + g * 16;
    v8bf lo = *(const v8bf*)(p);
    v8bf hi = *(const v8bf*)(p + 8);
#pragma unroll
    for (int i = 0; i < 8; ++i) { qa[ks][i] = lo[i]; qa[ks][i + 8] = hi[i]; }
  }

  v8f O[16];
#pragma unroll
  for (int dt = 0; dt < 16; ++dt)
#pragma unroll
    for (int r = 0; r < 8; ++r) O[dt][r] = 0.f;
  float mx[8], lsum[8];
#pragma unroll
  for (int r = 0; r < 8; ++r) { mx[r] = -__builtin_inff(); lsum[r] = 0.f; }

  for (int jp = 0; jp <= jt; jp += 2) {
    const int s0 = jp * 16;   // first tile; second tile s0+16 (masked by causality)

    // S pair: two 16x16 tiles of q @ k^T.
    v8f S0, S1;
#pragma unroll
    for (int r = 0; r < 8; ++r) { S0[r] = 0.f; S1[r] = 0.f; }
#pragma unroll
    for (int ks = 0; ks < 8; ++ks) {
      v16bf kf0 = *(const v16bf*)(kb + (size_t)(s0 + m) * Dn + ks * 32 + g * 16);
      v16bf kf1 = *(const v16bf*)(kb + (size_t)(s0 + 16 + m) * Dn + ks * 32 + g * 16);
      S0 = __builtin_amdgcn_wmma_f32_16x16x32_bf16(false, qa[ks], false, kf0,
                                                   (short)0, S0, false, false);
      S1 = __builtin_amdgcn_wmma_f32_16x16x32_bf16(false, qa[ks], false, kf1,
                                                   (short)0, S1, false, false);
    }

    const int col0 = s0 + m;
    const int col1 = s0 + 16 + m;
#pragma unroll
    for (int r = 0; r < 8; ++r) {
      const int trow = t0 + r + 8 * g;
      const bool ok0 = (col0 <= trow) && (col0 < len) && (S0[r] != 0.0f);
      const bool ok1 = (col1 <= trow) && (col1 < len) && (S1[r] != 0.0f);
      const float w0 = S0[r] * 0.0625f;  // 1/sqrt(256)
      const float w1 = S1[r] * 0.0625f;
      float x = fmaxf(ok0 ? w0 : -__builtin_inff(), ok1 ? w1 : -__builtin_inff());
      x = fmaxf(x, __shfl_xor(x, 1, 32));
      x = fmaxf(x, __shfl_xor(x, 2, 32));
      x = fmaxf(x, __shfl_xor(x, 4, 32));
      x = fmaxf(x, __shfl_xor(x, 8, 32));
      const float mnew = fmaxf(mx[r], x);
      const float pv0 = ok0 ? __expf(w0 - mnew) : 0.f;
      const float pv1 = ok1 ? __expf(w1 - mnew) : 0.f;
      float rs = pv0 + pv1;
      rs += __shfl_xor(rs, 1, 32);
      rs += __shfl_xor(rs, 2, 32);
      rs += __shfl_xor(rs, 4, 32);
      rs += __shfl_xor(rs, 8, 32);
      const float corr = (mx[r] == -__builtin_inff()) ? 0.f : __expf(mx[r] - mnew);
      lsum[r] = lsum[r] * corr + rs;
      mx[r] = mnew;
      sP[r + 8 * g][m]      = (bf16)pv0;
      sP[r + 8 * g][16 + m] = (bf16)pv1;
#pragma unroll
      for (int dt = 0; dt < 16; ++dt) O[dt][r] *= corr;
    }
    wave_lds_fence();

    // P pair-tile as a FULL K=32 A-fragment.
    v16bf pa;
    {
      const bf16* pr = &sP[m][g * 16];
#pragma unroll
      for (int i = 0; i < 8; ++i) { pa[i] = pr[i]; pa[i + 8] = pr[8 + i]; }
    }

    // O += P @ V  (B-fragment: K = g*16+i -> rows s0+g*16+i, all lanes active)
#pragma unroll
    for (int dt = 0; dt < 16; ++dt) {
      v16bf bv;
      const bf16* vp = vb + (size_t)(s0 + g * 16) * Dn + dt * 16 + m;
#pragma unroll
      for (int i = 0; i < 16; ++i) bv[i] = vp[(size_t)i * Dn];
      O[dt] = __builtin_amdgcn_wmma_f32_16x16x32_bf16(false, pa, false, bv,
                                                      (short)0, O[dt], false, false);
    }
    wave_lds_fence();  // sP reused next iteration
  }

#pragma unroll
  for (int r = 0; r < 8; ++r) {
    const float inv = (lsum[r] > 0.f) ? 1.f / lsum[r] : 0.f;
    const size_t rowg = (size_t)(row0 + r + 8 * g);
#pragma unroll
    for (int dt = 0; dt < 16; ++dt)
      out[rowg * Dn + dt * 16 + m] = O[dt][r] * inv;
  }
}

// ---------------------------------------------------------------------------
// Kernel D: vmean[b][d] = mean over t of v  (pad-row uniform-softmax output)
// ---------------------------------------------------------------------------
__global__ void vmean_kernel(const bf16* __restrict__ vb, float* __restrict__ vmean) {
  const int b = blockIdx.x, d = threadIdx.x;
  float s = 0.f;
  for (int t = 0; t < Tn; ++t)
    s += (float)vb[((size_t)b * Tn + t) * Dn + d];
  vmean[(size_t)b * Dn + d] = s * (1.f / Tn);
}

// Kernel E: overwrite pad rows (t >= len) with vmean.
__global__ void padrow_kernel(const int* __restrict__ input_len,
                              const float* __restrict__ vmean, float* __restrict__ out) {
  const int row = blockIdx.x;
  const int b = row / Tn, t = row % Tn;
  if (t >= input_len[b])
    out[(size_t)row * Dn + threadIdx.x] = vmean[(size_t)b * Dn + threadIdx.x];
}

// ---------------------------------------------------------------------------
extern "C" void kernel_launch(void* const* d_in, const int* in_sizes, int n_in,
                              void* d_out, int out_size, void* d_ws, size_t ws_size,
                              hipStream_t stream) {
  const float* query = (const float*)d_in[0];
  const float* value = (const float*)d_in[1];
  const float* aux   = (const float*)d_in[2];
  const int*   ilen  = (const int*)d_in[3];
  const float* Wql = (const float*)d_in[4];  const float* bql = (const float*)d_in[5];
  const float* Wq  = (const float*)d_in[6];  const float* bq  = (const float*)d_in[7];
  const float* aq  = (const float*)d_in[8];  const float* gq  = (const float*)d_in[9];
  const float* btq = (const float*)d_in[10];
  const float* Wk  = (const float*)d_in[11]; const float* bk  = (const float*)d_in[12];
  const float* ak  = (const float*)d_in[13]; const float* gk  = (const float*)d_in[14];
  const float* btk = (const float*)d_in[15];
  const float* Wv  = (const float*)d_in[16]; const float* bv  = (const float*)d_in[17];
  const float* av  = (const float*)d_in[18]; const float* gv  = (const float*)d_in[19];
  const float* btv = (const float*)d_in[20];
  const float* Ws  = (const float*)d_in[21]; const float* bs  = (const float*)d_in[22];
  const float* Wt  = (const float*)d_in[23]; const float* bt  = (const float*)d_in[24];
  float* out = (float*)d_out;

  // workspace layout
  char* ws = (char*)d_ws;
  bf16* qb = (bf16*)ws;                                   // 8 MB
  bf16* kb = qb + (size_t)NROW * Dn;                      // 8 MB
  bf16* vb = kb + (size_t)NROW * Dn;                      // 8 MB
  bf16* WT = vb + (size_t)NROW * Dn;                      // 6 * 128 KB
  float* auxb  = (float*)(WT + 6 * 65536);                // 3*B*D f32
  float* vmean = auxb + 3 * Bn * Dn;                      // B*D f32

  prep_weights<<<dim3(256, 6), 256, 0, stream>>>(Wql, Wq, Wk, Wv, Ws, Wt, WT);
  prep_auxbias<<<dim3(Bn, 3), 256, 0, stream>>>(aux, Wq, bq, Wk, bk, Wv, bv, auxb);
  proj_kernel<<<NROW / 16, 32, 0, stream>>>(query, value, WT, auxb, bql,
                                            aq, gq, btq, ak, gk, btk, av, gv, btv,
                                            bs, bt, qb, kb, vb);
  vmean_kernel<<<Bn, 256, 0, stream>>>(vb, vmean);
  attn_kernel<<<NROW / 16, 32, 0, stream>>>(qb, kb, vb, ilen, out);
  padrow_kernel<<<NROW, 256, 0, stream>>>(ilen, vmean, out);
}
